// TypeAreaScaledDotProductAttention_Big2_84628035600749
// MI455X (gfx1250) — matrix operand order repaired and checked by reading, not compiled, not run on hardware
//
#include <hip/hip_runtime.h>
#include <hip/hip_bf16.h>

// ---------------------------------------------------------------------------
// TypeAreaScaledDotProductAttention (Big2): fused attention on MI455X gfx1250.
//   Kernel 1: q_sum/k_sum -> bf16 in workspace.
//   Kernel 2: flash-style attention, v_wmma_f32_16x16x32_bf16 for both GEMMs,
//             V staged global->LDS with GLOBAL_LOAD_ASYNC_TO_LDS_B128
//             (double-buffered, shared by all 4 waves of the WG).
// Memory-bound: ~600 MB of HBM traffic -> ~26 us ideal at 23.3 TB/s.
// ---------------------------------------------------------------------------

#define BB 16
#define HH 8
#define SS 512
#define DD 64
#define NV 9
#define INV_TEMP 0.125f
#define NEG_INF (-1e9f)

typedef __attribute__((ext_vector_type(16))) __bf16 v16bf;
typedef __attribute__((ext_vector_type(16))) unsigned short v16us;
typedef __attribute__((ext_vector_type(8)))  float v8f;

struct Ptrs9 { const float* p[NV]; };

__device__ __forceinline__ unsigned short f32_to_bf16u(float f) {
  unsigned int u = __float_as_uint(f);
  unsigned int r = u + 0x7FFFu + ((u >> 16) & 1u);   // round-to-nearest-even
  return (unsigned short)(r >> 16);
}

// ---------------------------------------------------------------------------
// Kernel 1: qsum/ksum = sum of 9 tensors, f32 -> bf16 into workspace
// ---------------------------------------------------------------------------
__global__ void sum_qk_kernel(Ptrs9 q, Ptrs9 k,
                              unsigned short* __restrict__ qs,
                              unsigned short* __restrict__ ks, int n) {
  int i = blockIdx.x * blockDim.x + threadIdx.x;
  if (i >= n) return;
  float aq = 0.f, ak = 0.f;
#pragma unroll
  for (int t = 0; t < NV; ++t) { aq += q.p[t][i]; ak += k.p[t][i]; }
  qs[i] = f32_to_bf16u(aq);
  ks[i] = f32_to_bf16u(ak);
}

// ---------------------------------------------------------------------------
// Kernel 2: fused attention. 4 waves/WG, 16 q-rows per wave, 64 rows per WG.
// grid.x = B*H*(S/64) = 1024
// Dynamic LDS: 4*(16*512*4) scores (reused as V staging in phase 3)
//            + 4*(16*512*2) P (bf16) + 4*16*4 rowinv  = 196,864 B
// ---------------------------------------------------------------------------
#define WAVES 4
#define LDS_BYTES (WAVES * 16 * SS * 4 + WAVES * 16 * SS * 2 + WAVES * 16 * 4)
#define VCHUNK (32 * DD)                 // one V chunk: 32 rows x 64 cols f32

extern __shared__ char smem_raw[];

__global__ void __launch_bounds__(WAVES * 32)
attn_kernel(const unsigned short* __restrict__ qsum,
            const unsigned short* __restrict__ ksum,
            Ptrs9 v, const int* __restrict__ mask,
            float* __restrict__ out) {
  const int tid  = threadIdx.x;
  const int wave = tid >> 5;
  const int lane = tid & 31;
  const int hl   = lane >> 4;     // lane half: 0 or 1
  const int l16  = lane & 15;

  const int wg = blockIdx.x;          // 0..1023
  const int qt = wg & 7;              // q-tile within (b,h), S/64 = 8
  const int bh = wg >> 3;             // 0..127
  const int b  = bh >> 3;             // H = 8

  const int mrow0 = qt * 64 + wave * 16;               // first q row of wave
  const unsigned short* qbase = qsum + (size_t)bh * SS * DD;
  const unsigned short* kbase = ksum + (size_t)bh * SS * DD;
  const int* mbase = mask + (size_t)b * SS * SS;

  float*          sc     = (float*)smem_raw + (size_t)wave * 16 * SS;
  unsigned short* pp     = (unsigned short*)(smem_raw + WAVES * 16 * SS * 4)
                           + (size_t)wave * 16 * SS;
  float*          rowinv = (float*)(smem_raw + WAVES * 16 * SS * 4
                                             + WAVES * 16 * SS * 2)
                           + wave * 16;
  // Phase-3 V staging reuses the (dead) score region: 2 x 8 KB double buffer
  float* vstage = (float*)smem_raw;

  // ---- load Q A-fragments (16x64 bf16 -> two 16x32 frags), ISA A layout ----
  v16us a_us0, a_us1;
  {
    const unsigned short* qp = qbase + (size_t)(mrow0 + l16) * DD;
#pragma unroll
    for (int i = 0; i < 16; ++i) {
      int kk = ((i >> 3) << 4) + hl * 8 + (i & 7);     // K within 32-chunk
      a_us0[i] = qp[kk];
      a_us1[i] = qp[32 + kk];
    }
  }
  const v16bf a0 = __builtin_bit_cast(v16bf, a_us0);
  const v16bf a1 = __builtin_bit_cast(v16bf, a_us1);

  // ---- Phase 1: scores = Qsum @ Ksum^T, 32 key tiles of 16 ----
  for (int nt = 0; nt < SS / 16; ++nt) {
    const unsigned short* kp = kbase + (size_t)(nt * 16 + l16) * DD;
    v16us b_us0, b_us1;
#pragma unroll
    for (int i = 0; i < 16; ++i) {                     // B layout: K = ((i>>3)<<4) + hl*8 + (i&7)
      b_us0[i] = kp[((i >> 3) << 4) + hl * 8 + (i & 7)];
      b_us1[i] = kp[32 + ((i >> 3) << 4) + hl * 8 + (i & 7)];
    }
    v8f c = {0.f, 0.f, 0.f, 0.f, 0.f, 0.f, 0.f, 0.f};
    c = __builtin_amdgcn_wmma_f32_16x16x32_bf16(
        false, a0, false, __builtin_bit_cast(v16bf, b_us0), (short)0, c, false, false);
    c = __builtin_amdgcn_wmma_f32_16x16x32_bf16(
        false, a1, false, __builtin_bit_cast(v16bf, b_us1), (short)0, c, false, false);
    // C layout: element j -> row (j + 8*hl), col (nt*16 + l16)
#pragma unroll
    for (int j = 0; j < 8; ++j)
      sc[(j + 8 * hl) * SS + nt * 16 + l16] = c[j];
  }
  __syncthreads();

  // ---- Phase 2: scale + mask + row softmax (2 lanes per row) ----
  {
    const int r = l16;
    const int cbase = hl * (SS / 2);
    float* srow = sc + r * SS;
    const int* mrow = mbase + (size_t)(mrow0 + r) * SS;
    unsigned short* prow = pp + r * SS;

    float mx = -3.4e38f;
    for (int c0 = 0; c0 < SS / 2; ++c0) {
      int col = cbase + c0;
      float s = srow[col] * INV_TEMP;
      s = (mrow[col] == 0) ? NEG_INF : s;
      srow[col] = s;
      mx = fmaxf(mx, s);
    }
    mx = fmaxf(mx, __shfl_xor(mx, 16, 32));

    float sum = 0.f;
    for (int c0 = 0; c0 < SS / 2; ++c0) {
      int col = cbase + c0;
      float e = __expf(srow[col] - mx);
      sum += e;
      prow[col] = f32_to_bf16u(e);       // unnormalized P in bf16
    }
    sum += __shfl_xor(sum, 16, 32);
    if (hl == 0) rowinv[r] = 1.0f / sum;
  }
  __syncthreads();   // scores region is dead from here -> reused for V staging

  float ri[8];
#pragma unroll
  for (int j = 0; j < 8; ++j) ri[j] = rowinv[j + 8 * hl];

  // ---- Phase 3: out[n] = (P @ V_n) * rowinv, V staged async into LDS ----
  // Chunk c (0..143): n = c/16, kt = c%16, covers V_n rows [kt*32, kt*32+32).
  // Each chunk = 8 KB f32, staged once per WG via async b128 copies
  // (128 threads x 4 x 16B), double-buffered in the dead score region.
  auto stage_chunk = [&](int c, int buf) {
    const int n  = c >> 4;
    const int kt = c & 15;
    const float* vb = v.p[n] + (size_t)bh * SS * DD + (size_t)kt * 32 * DD;
    unsigned ldsbase = (unsigned)(uintptr_t)(vstage + buf * VCHUNK);
#pragma unroll
    for (int j = 0; j < 4; ++j) {
      int e = (j * (WAVES * 32) + tid) * 4;            // float index in chunk
      unsigned lofs = ldsbase + (unsigned)(e * 4);
      unsigned long long ga = (unsigned long long)(uintptr_t)(vb + e);
      asm volatile("global_load_async_to_lds_b128 %0, %1, off"
                   :: "v"(lofs), "v"(ga) : "memory");
    }
  };

  stage_chunk(0, 0);
  asm volatile("s_wait_asynccnt 0x0" ::: "memory");
  __syncthreads();

  int c = 0;
  for (int n = 0; n < NV; ++n) {
    v8f acc0 = {0.f,0.f,0.f,0.f,0.f,0.f,0.f,0.f};
    v8f acc1 = acc0, acc2 = acc0, acc3 = acc0;

    for (int kt = 0; kt < SS / 32; ++kt, ++c) {
      const int buf = c & 1;
      if (c + 1 < NV * (SS / 32)) stage_chunk(c + 1, buf ^ 1);

      // A frag of P from LDS (16x32 bf16), ISA A layout
      v16us pa;
      const unsigned short* pr = pp + l16 * SS + kt * 32;
#pragma unroll
      for (int i = 0; i < 16; ++i)
        pa[i] = pr[((i >> 3) << 4) + hl * 8 + (i & 7)];
      const v16bf A = __builtin_bit_cast(v16bf, pa);

      // B frags from staged V chunk: col (nt*16 + l16), K = ((i>>3)<<4) + hl*8 + (i&7)
      const float* vs = vstage + buf * VCHUNK;
#pragma unroll
      for (int nt = 0; nt < 4; ++nt) {
        v16us bb;
        const float* vc = vs + nt * 16 + l16;
#pragma unroll
        for (int i = 0; i < 16; ++i)
          bb[i] = f32_to_bf16u(vc[(((i >> 3) << 4) + hl * 8 + (i & 7)) * DD]);
        v8f* accp = (nt == 0) ? &acc0 : (nt == 1) ? &acc1 : (nt == 2) ? &acc2 : &acc3;
        *accp = __builtin_amdgcn_wmma_f32_16x16x32_bf16(
            false, A, false, __builtin_bit_cast(v16bf, bb), (short)0, *accp, false, false);
      }

      // next buffer fully landed in LDS before anyone reads it
      asm volatile("s_wait_asynccnt 0x0" ::: "memory");
      __syncthreads();
    }

    float* ob = out + ((size_t)n * BB * HH + bh) * SS * DD + (size_t)mrow0 * DD;
#pragma unroll
    for (int j = 0; j < 8; ++j) {
      int m = j + 8 * hl;
      ob[(size_t)m * DD +  0 + l16] = acc0[j] * ri[j];
      ob[(size_t)m * DD + 16 + l16] = acc1[j] * ri[j];
      ob[(size_t)m * DD + 32 + l16] = acc2[j] * ri[j];
      ob[(size_t)m * DD + 48 + l16] = acc3[j] * ri[j];
    }
  }
}

// ---------------------------------------------------------------------------
extern "C" void kernel_launch(void* const* d_in, const int* in_sizes, int n_in,
                              void* d_out, int out_size, void* d_ws, size_t ws_size,
                              hipStream_t stream) {
  (void)in_sizes; (void)n_in; (void)out_size; (void)ws_size;

  Ptrs9 q, k, v;
  for (int t = 0; t < NV; ++t) {
    q.p[t] = (const float*)d_in[3 * t + 0];
    k.p[t] = (const float*)d_in[3 * t + 1];
    v.p[t] = (const float*)d_in[3 * t + 2];
  }
  const int* mask = (const int*)d_in[27];
  float* out = (float*)d_out;

  const int N = BB * HH * SS * DD;                       // 4,194,304 elements
  unsigned short* qsum = (unsigned short*)d_ws;          // 8 MB
  unsigned short* ksum = qsum + N;                       // 8 MB

  sum_qk_kernel<<<(N + 255) / 256, 256, 0, stream>>>(q, k, qsum, ksum, N);

  const int grid = BB * HH * (SS / 64);                  // 1024 workgroups
  attn_kernel<<<grid, WAVES * 32, LDS_BYTES, stream>>>(qsum, ksum, v, mask, out);
}
